// GAT_25821343384334
// MI455X (gfx1250) — matrix operand order repaired and checked by reading, not compiled, not run on hardware
//
#include <hip/hip_runtime.h>
#include <hip/hip_bf16.h>
#include <cmath>

#define SLOPE 0.2f
#define FIN 256
#define HID 64

typedef __attribute__((ext_vector_type(16))) _Float16 v16h;
typedef __attribute__((ext_vector_type(8)))  float    v8f;

__device__ __forceinline__ float atomicMaxF(float* addr, float v) {
  if (v >= 0.f) return __int_as_float(atomicMax((int*)addr, __float_as_int(v)));
  return __uint_as_float(atomicMin((unsigned int*)addr, __float_as_uint(v)));
}

// ---------------- utility ----------------
__global__ void k_fill(float* __restrict__ p, float v, int n) {
  int i = blockIdx.x * blockDim.x + threadIdx.x;
  if (i < n) p[i] = v;
}

// vectorized f32 -> f16 convert, 8 elements per thread (n must be a multiple of 8)
__global__ void k_cvt_f2h_v8(const float* __restrict__ in, _Float16* __restrict__ out, int n8) {
  int i = blockIdx.x * blockDim.x + threadIdx.x;
  if (i >= n8) return;
  const float4* ip = (const float4*)(in + (size_t)i * 8);
  float4 a = ip[0], b = ip[1];
  _Float16 h[8] = {(_Float16)a.x, (_Float16)a.y, (_Float16)a.z, (_Float16)a.w,
                   (_Float16)b.x, (_Float16)b.y, (_Float16)b.z, (_Float16)b.w};
  *(uint4*)(out + (size_t)i * 8) = *(const uint4*)h;
}

// convert + transpose weights: W[K,NC] f32 row-major -> Wt[NC,K] f16
__global__ void k_cvt_f2h_T(const float* __restrict__ W, _Float16* __restrict__ Wt,
                            int K, int NC) {
  int t = blockIdx.x * blockDim.x + threadIdx.x;   // over NC*K, coalesced writes
  if (t >= K * NC) return;
  int n = t / K, k = t - n * K;
  Wt[t] = (_Float16)W[(size_t)k * NC + n];
}

// ---------------- WMMA GEMM: C[M,NC=64] = A[M,K](f16) @ B (Bt[NC,K] f16), f32 accum ----
// block = 256 threads = 8 waves; each wave owns a 16x64 strip (4 x 16x16 tiles).
__global__ void k_gemm_wmma(const _Float16* __restrict__ A, const _Float16* __restrict__ Bt,
                            float* __restrict__ C, int M, int K) {
  const int wave = threadIdx.x >> 5;
  const int lane = threadIdx.x & 31;
  const int rowBase = (blockIdx.x * 8 + wave) * 16;
  const int m  = lane & 15;
  const int ks = lane >> 4;                 // 0/1: K-half select (this code's contiguous k order, same for A and B)
  int arow = rowBase + m;
  if (arow >= M) arow = M - 1;              // clamp (stores guarded below)
  const _Float16* Ap = A + (size_t)arow * K + ks * 16;
  v8f acc0 = {}, acc1 = {}, acc2 = {}, acc3 = {};
  for (int k0 = 0; k0 < K; k0 += 32) {
    v16h a;
    // A 16x32 layout: lanes 0-15: K=[k0..k0+15]; lanes 16-31: K=[k0+16..k0+31]
    ((uint4*)&a)[0] = *(const uint4*)(Ap + k0);
    ((uint4*)&a)[1] = *(const uint4*)(Ap + k0 + 8);
    const int kb = k0 + ks * 16;            // B: 16 consecutive K values for column n
    v16h b0, b1, b2, b3;
    {
      const _Float16* Bp = Bt + (size_t)(0 * 16 + m) * K + kb;
      ((uint4*)&b0)[0] = *(const uint4*)(Bp);
      ((uint4*)&b0)[1] = *(const uint4*)(Bp + 8);
    }
    {
      const _Float16* Bp = Bt + (size_t)(1 * 16 + m) * K + kb;
      ((uint4*)&b1)[0] = *(const uint4*)(Bp);
      ((uint4*)&b1)[1] = *(const uint4*)(Bp + 8);
    }
    {
      const _Float16* Bp = Bt + (size_t)(2 * 16 + m) * K + kb;
      ((uint4*)&b2)[0] = *(const uint4*)(Bp);
      ((uint4*)&b2)[1] = *(const uint4*)(Bp + 8);
    }
    {
      const _Float16* Bp = Bt + (size_t)(3 * 16 + m) * K + kb;
      ((uint4*)&b3)[0] = *(const uint4*)(Bp);
      ((uint4*)&b3)[1] = *(const uint4*)(Bp + 8);
    }
    acc0 = __builtin_amdgcn_wmma_f32_16x16x32_f16(false, a, false, b0, (short)0, acc0, false, false);
    acc1 = __builtin_amdgcn_wmma_f32_16x16x32_f16(false, a, false, b1, (short)0, acc1, false, false);
    acc2 = __builtin_amdgcn_wmma_f32_16x16x32_f16(false, a, false, b2, (short)0, acc2, false, false);
    acc3 = __builtin_amdgcn_wmma_f32_16x16x32_f16(false, a, false, b3, (short)0, acc3, false, false);
  }
  // D layout: VGPR r -> row rowBase + r + ks*8, column = tile*16 + m
  float* Cw = C + (size_t)(rowBase + ks * 8) * HID + m;
  if (rowBase + 16 <= M) {
#pragma unroll
    for (int r = 0; r < 8; ++r) {
      float* cr = Cw + (size_t)r * HID;
      cr[0] = acc0[r]; cr[16] = acc1[r]; cr[32] = acc2[r]; cr[48] = acc3[r];
    }
  } else {
#pragma unroll
    for (int r = 0; r < 8; ++r) {
      if (rowBase + r + ks * 8 < M) {
        float* cr = Cw + (size_t)r * HID;
        cr[0] = acc0[r]; cr[16] = acc1[r]; cr[32] = acc2[r]; cr[48] = acc3[r];
      }
    }
  }
}

// ---------------- attention projections ----------------
__global__ void k_proj8(const float* __restrict__ h, const float* __restrict__ as,
                        const float* __restrict__ ad, float* __restrict__ als,
                        float* __restrict__ ald, int NH) {
  int t = blockIdx.x * blockDim.x + threadIdx.x;
  if (t >= NH) return;
  int hd = t & 7;
  const float4* hp = (const float4*)(h + (size_t)t * 8);   // == h[n*64 + hd*8]
  const float4* sp = (const float4*)(as + hd * 8);
  const float4* dp = (const float4*)(ad + hd * 8);
  float4 h0 = hp[0], h1 = hp[1];
  float4 s0 = sp[0], s1 = sp[1];
  float4 d0 = dp[0], d1 = dp[1];
  als[t] = h0.x*s0.x + h0.y*s0.y + h0.z*s0.z + h0.w*s0.w +
           h1.x*s1.x + h1.y*s1.y + h1.z*s1.z + h1.w*s1.w;
  ald[t] = h0.x*d0.x + h0.y*d0.y + h0.z*d0.z + h0.w*d0.w +
           h1.x*d1.x + h1.y*d1.y + h1.z*d1.z + h1.w*d1.w;
}

__global__ void k_proj64(const float* __restrict__ h, const float* __restrict__ as,
                         const float* __restrict__ ad, float* __restrict__ als,
                         float* __restrict__ ald, int N) {
  int n = blockIdx.x * blockDim.x + threadIdx.x;
  if (n >= N) return;
  const float4* hp = (const float4*)(h + (size_t)n * 64);
  const float4* sp = (const float4*)as;
  const float4* dp = (const float4*)ad;
  float ss = 0.f, sd = 0.f;
#pragma unroll
  for (int i = 0; i < 16; ++i) {
    float4 hv = hp[i], sv = sp[i], dv = dp[i];
    ss += hv.x*sv.x + hv.y*sv.y + hv.z*sv.z + hv.w*sv.w;
    sd += hv.x*dv.x + hv.y*dv.y + hv.z*dv.z + hv.w*dv.w;
  }
  als[n] = ss;
  ald[n] = sd;
}

// ---------------- edge kernels ----------------
__device__ __forceinline__ void edge_info(const int* __restrict__ src,
                                          const int* __restrict__ dst,
                                          const float* __restrict__ ew,
                                          int E, int e, int& s, int& d, float& corr) {
  if (e < E) { s = src[e]; d = dst[e]; corr = 1.f - 1.f / ew[e]; }
  else       { s = d = e - E; corr = 0.f; }   // self-loop, ew=1 -> corr=0
}

__global__ void k_edge_max8(const int* __restrict__ src, const int* __restrict__ dst,
                            const float* __restrict__ ew, int E, int Etot,
                            const float* __restrict__ als, const float* __restrict__ ald,
                            float* __restrict__ mx) {
  int e = blockIdx.x * blockDim.x + threadIdx.x;
  if (e >= Etot) return;
  int s, d; float corr;
  edge_info(src, dst, ew, E, e, s, d, corr);
  const float4* sp = (const float4*)(als + (size_t)s * 8);
  const float4* dp = (const float4*)(ald + (size_t)d * 8);
  float4 a0 = sp[0], a1 = sp[1], b0 = dp[0], b1 = dp[1];
  float v[8] = {a0.x+b0.x, a0.y+b0.y, a0.z+b0.z, a0.w+b0.w,
                a1.x+b1.x, a1.y+b1.y, a1.z+b1.z, a1.w+b1.w};
#pragma unroll
  for (int h = 0; h < 8; ++h) {
    float xv = v[h];
    xv = (xv >= 0.f) ? xv : SLOPE * xv;
    atomicMaxF(&mx[(size_t)d * 8 + h], xv + corr);
  }
}

__global__ void k_edge_acc8(const int* __restrict__ src, const int* __restrict__ dst,
                            const float* __restrict__ ew, int E, int Etot,
                            const float* __restrict__ als, const float* __restrict__ ald,
                            const float* __restrict__ mx, const float* __restrict__ hsrc,
                            float* __restrict__ ssum, float* __restrict__ out) {
  int t = blockIdx.x * blockDim.x + threadIdx.x;
  if (t >= Etot * 8) return;
  int e = t >> 3, hd = t & 7;
  int s, d; float corr;
  edge_info(src, dst, ew, E, e, s, d, corr);
  float xv = als[(size_t)s * 8 + hd] + ald[(size_t)d * 8 + hd];
  xv = (xv >= 0.f) ? xv : SLOPE * xv;
  xv += corr;
  float ee = expf(xv - mx[(size_t)d * 8 + hd]);
  atomicAdd(&ssum[(size_t)d * 8 + hd], ee);
  const float4* hp = (const float4*)(hsrc + (size_t)s * 64 + hd * 8);
  float4 h0 = hp[0], h1 = hp[1];
  float* op = out + (size_t)d * 64 + hd * 8;
  atomicAdd(op + 0, ee * h0.x); atomicAdd(op + 1, ee * h0.y);
  atomicAdd(op + 2, ee * h0.z); atomicAdd(op + 3, ee * h0.w);
  atomicAdd(op + 4, ee * h1.x); atomicAdd(op + 5, ee * h1.y);
  atomicAdd(op + 6, ee * h1.z); atomicAdd(op + 7, ee * h1.w);
}

__global__ void k_edge_max1(const int* __restrict__ src, const int* __restrict__ dst,
                            const float* __restrict__ ew, int E, int Etot,
                            const float* __restrict__ als, const float* __restrict__ ald,
                            float* __restrict__ mx) {
  int e = blockIdx.x * blockDim.x + threadIdx.x;
  if (e >= Etot) return;
  int s, d; float corr;
  edge_info(src, dst, ew, E, e, s, d, corr);
  float xv = als[s] + ald[d];
  xv = (xv >= 0.f) ? xv : SLOPE * xv;
  atomicMaxF(&mx[d], xv + corr);
}

__global__ void k_edge_acc1(const int* __restrict__ src, const int* __restrict__ dst,
                            const float* __restrict__ ew, int E, int Etot,
                            const float* __restrict__ als, const float* __restrict__ ald,
                            const float* __restrict__ mx, const float* __restrict__ hsrc,
                            float* __restrict__ ssum, float* __restrict__ out) {
  int t = blockIdx.x * blockDim.x + threadIdx.x;
  if (t >= Etot * 8) return;
  int e = t >> 3, q = t & 7;
  int s, d; float corr;
  edge_info(src, dst, ew, E, e, s, d, corr);
  float xv = als[s] + ald[d];
  xv = (xv >= 0.f) ? xv : SLOPE * xv;
  xv += corr;
  float ee = expf(xv - mx[d]);
  if (q == 0) atomicAdd(&ssum[d], ee);
  const float4* hp = (const float4*)(hsrc + (size_t)s * 64 + q * 8);
  float4 h0 = hp[0], h1 = hp[1];
  float* op = out + (size_t)d * 64 + q * 8;
  atomicAdd(op + 0, ee * h0.x); atomicAdd(op + 1, ee * h0.y);
  atomicAdd(op + 2, ee * h0.z); atomicAdd(op + 3, ee * h0.w);
  atomicAdd(op + 4, ee * h1.x); atomicAdd(op + 5, ee * h1.y);
  atomicAdd(op + 6, ee * h1.z); atomicAdd(op + 7, ee * h1.w);
}

// ---------------- finalize ----------------
__global__ void k_fin1(const float* __restrict__ out1, const float* __restrict__ s1,
                       const float* __restrict__ b1, _Float16* __restrict__ xh2, int total) {
  int t = blockIdx.x * blockDim.x + threadIdx.x;
  if (t >= total) return;
  int n = t >> 6, k = t & 63;
  float v = out1[t] / (s1[(size_t)n * 8 + (k >> 3)] + 1e-16f) + b1[k];
  v = (v > 0.f) ? v : (expf(v) - 1.f);   // ELU
  xh2[t] = (_Float16)v;
}

__global__ void k_fin2(float* __restrict__ out, const float* __restrict__ s2,
                       const float* __restrict__ b2, int total) {
  int t = blockIdx.x * blockDim.x + threadIdx.x;
  if (t >= total) return;
  int n = t >> 6, k = t & 63;
  out[t] = out[t] / (s2[n] + 1e-16f) + b2[k];
}

extern "C" void kernel_launch(void* const* d_in, const int* in_sizes, int n_in,
                              void* d_out, int out_size, void* d_ws, size_t ws_size,
                              hipStream_t stream) {
  const float* x   = (const float*)d_in[0];
  const int*   ei  = (const int*)d_in[1];
  const float* ew  = (const float*)d_in[2];
  const float* W1  = (const float*)d_in[3];
  const float* as1 = (const float*)d_in[4];
  const float* ad1 = (const float*)d_in[5];
  const float* b1  = (const float*)d_in[6];
  const float* W2  = (const float*)d_in[7];
  const float* as2 = (const float*)d_in[8];
  const float* ad2 = (const float*)d_in[9];
  const float* b2  = (const float*)d_in[10];

  const int N = in_sizes[0] / FIN;
  const int E = in_sizes[1] / 2;
  const int* src = ei;
  const int* dst = ei + E;
  const int Etot = E + N;
  float* out = (float*)d_out;

  // workspace layout
  char* ws = (char*)d_ws;
  size_t off = 0;
  auto alloc = [&](size_t bytes) -> char* {
    char* p = ws + off;
    off = (off + bytes + 255) & ~(size_t)255;
    return p;
  };
  _Float16* XH   = (_Float16*)alloc((size_t)N * FIN * sizeof(_Float16)); // phase-1 f16 input
  float*    H1   = (float*)   alloc((size_t)N * HID * sizeof(float));
  float*    ALS1 = (float*)   alloc((size_t)N * 8 * sizeof(float));
  float*    ALD1 = (float*)   alloc((size_t)N * 8 * sizeof(float));
  float*    M1   = (float*)   alloc((size_t)N * 8 * sizeof(float));
  float*    S1   = (float*)   alloc((size_t)N * 8 * sizeof(float));
  float*    OUT1 = (float*)   alloc((size_t)N * HID * sizeof(float));
  _Float16* XH2  = (_Float16*)alloc((size_t)N * HID * sizeof(_Float16));
  _Float16* W1T  = (_Float16*)alloc((size_t)HID * FIN * sizeof(_Float16)); // [NC,K] col-major weights
  _Float16* W2T  = (_Float16*)alloc((size_t)HID * HID * sizeof(_Float16));
  // phase-2 buffers overlay the (now dead) XH region
  float* H2   = (float*)XH;
  float* ALS2 = (float*)((char*)XH + (size_t)N * HID * sizeof(float));
  float* ALD2 = ALS2 + N;
  float* M2   = ALD2 + N;
  float* S2   = M2 + N;

  const int TB = 256;
  auto G = [](long n, int b) { return (int)((n + b - 1) / b); };

  // ---- layer 1 ----
  k_cvt_f2h_v8<<<G((long)N * FIN / 8, TB), TB, 0, stream>>>(x, XH, N * FIN / 8);
  k_cvt_f2h_T<<<G(FIN * HID, TB), TB, 0, stream>>>(W1, W1T, FIN, HID);
  k_cvt_f2h_T<<<G(HID * HID, TB), TB, 0, stream>>>(W2, W2T, HID, HID);

  k_gemm_wmma<<<G(N, 128), 256, 0, stream>>>(XH, W1T, H1, N, FIN);
  k_proj8<<<G((long)N * 8, TB), TB, 0, stream>>>(H1, as1, ad1, ALS1, ALD1, N * 8);

  k_fill<<<G((long)N * 8, TB), TB, 0, stream>>>(M1, -INFINITY, N * 8);
  k_fill<<<G((long)N * 8, TB), TB, 0, stream>>>(S1, 0.f, N * 8);
  k_fill<<<G((long)N * HID, TB), TB, 0, stream>>>(OUT1, 0.f, N * HID);

  k_edge_max8<<<G(Etot, TB), TB, 0, stream>>>(src, dst, ew, E, Etot, ALS1, ALD1, M1);
  k_edge_acc8<<<G((long)Etot * 8, TB), TB, 0, stream>>>(src, dst, ew, E, Etot,
                                                        ALS1, ALD1, M1, H1, S1, OUT1);
  k_fin1<<<G((long)N * HID, TB), TB, 0, stream>>>(OUT1, S1, b1, XH2, N * HID);

  // ---- layer 2 ----
  k_gemm_wmma<<<G(N, 128), 256, 0, stream>>>(XH2, W2T, H2, N, HID);
  k_proj64<<<G(N, TB), TB, 0, stream>>>(H2, as2, ad2, ALS2, ALD2, N);

  k_fill<<<G(N, TB), TB, 0, stream>>>(M2, -INFINITY, N);
  k_fill<<<G(N, TB), TB, 0, stream>>>(S2, 0.f, N);
  k_fill<<<G((long)N * HID, TB), TB, 0, stream>>>(out, 0.f, N * HID);

  k_edge_max1<<<G(Etot, TB), TB, 0, stream>>>(src, dst, ew, E, Etot, ALS2, ALD2, M2);
  k_edge_acc1<<<G((long)Etot * 8, TB), TB, 0, stream>>>(src, dst, ew, E, Etot,
                                                        ALS2, ALD2, M2, H2, S2, out);
  k_fin2<<<G((long)N * HID, TB), TB, 0, stream>>>(out, S2, b2, N * HID);
}
